// Attention_89197880803342
// MI455X (gfx1250) — matrix operand order repaired and checked by reading, not compiled, not run on hardware
//
#include <hip/hip_runtime.h>

// ---------------------------------------------------------------------------
// MI455X (gfx1250, wave32) RGBT local/global attention block.
// Channel-mixing GEMMs on v_wmma_f32_16x16x32_bf16, 128x128 block tiles,
// double-buffered LDS software pipeline, bf16 [N][K] weight cache.
// ---------------------------------------------------------------------------

typedef __attribute__((ext_vector_type(16))) __bf16 v16bf;
typedef __attribute__((ext_vector_type(8)))  __bf16 v8bf;
typedef __attribute__((ext_vector_type(8)))  float  v8f;

#define BM 128
#define BN 128
#define BK 32
#define LDP (BK + 8)   // LDS pitch in halves (80B rows, 16B aligned)

// round-half-up bf16 pair pack: pure 32-bit ALU, no half-register ops
__device__ __forceinline__ unsigned pack2bf(float lo, float hi) {
  return ((__float_as_uint(hi) + 0x8000u) & 0xFFFF0000u) |
         ((__float_as_uint(lo) + 0x8000u) >> 16);
}

// ---------------------------- LayerNorm ------------------------------------
__global__ void ln_kernel(const float* __restrict__ x, const float* __restrict__ w,
                          const float* __restrict__ b, float* __restrict__ y, int C) {
  __shared__ float sm[256];
  int t = blockIdx.x, c = threadIdx.x;
  float v = x[(size_t)t * C + c];
  sm[c] = v; __syncthreads();
  for (int s = C >> 1; s > 0; s >>= 1) { if (c < s) sm[c] += sm[c + s]; __syncthreads(); }
  float mean = sm[0] * (1.0f / C);
  __syncthreads();
  float d = v - mean;
  sm[c] = d * d; __syncthreads();
  for (int s = C >> 1; s > 0; s >>= 1) { if (c < s) sm[c] += sm[c + s]; __syncthreads(); }
  float var = sm[0] * (1.0f / C);
  y[(size_t)t * C + c] = d * rsqrtf(var + 1e-6f) * w[c] + b[c];
}

// -------- weight convert+transpose: Wt[n][k] = bf16(W[k][n]) ---------------
__global__ void wconv_kernel(const float* __restrict__ W, unsigned* __restrict__ Wt,
                             int K, int N) {
  int Kh = K >> 1;
  int idx = blockIdx.x * 256 + threadIdx.x;   // over N*Kh packed pairs
  if (idx >= N * Kh) return;
  int n = idx / Kh, k0 = (idx % Kh) * 2;
  Wt[idx] = pack2bf(W[(size_t)k0 * N + n], W[(size_t)(k0 + 1) * N + n]);
}

// ---------------------- WMMA GEMM pipeline step ----------------------------
// One k-step: sync, issue next-tile global loads, compute 8 WMMAs from buffer
// CUR, then commit staged registers into buffer CUR^1.
template <bool GATED, bool BOUNDS, int CUR>
__device__ __forceinline__
void gemm_step(const float* __restrict__ A, int lda,
               const __bf16* __restrict__ Wt,
               const float* __restrict__ gate, int rows_per_batch,
               int M, int K, int it, int nk, int m0, int n0,
               int sr, int sc, int waveM, int waveN, int laneLo, int laneHi,
               __bf16 (&As)[2][BM][LDP], __bf16 (&Bs)[2][BN][LDP],
               v8f (&acc)[2][4]) {
  __syncthreads();
  const bool have = __builtin_expect(it + 1 < nk, 1);
  const int kn = (it + 1) * BK;
  float4 fa[2][2], fg[2][2];
  v8bf fb[2];
  bool okA[2];
  if (have) {
    #pragma unroll
    for (int j = 0; j < 2; ++j) {
      int r = j * 64 + sr;
      int gm = m0 + r;
      okA[j] = (!BOUNDS || gm < M);
      if (okA[j]) {
        const float* ap = A + (size_t)gm * lda + kn + sc;
        fa[j][0] = *(const float4*)ap;
        fa[j][1] = *(const float4*)(ap + 4);
        __builtin_prefetch(ap + BK, 0, 3);
        if (GATED) {
          const float* gp = gate + (size_t)(gm / rows_per_batch) * K + kn + sc;
          fg[j][0] = *(const float4*)gp;
          fg[j][1] = *(const float4*)(gp + 4);
        }
      }
      fb[j] = *(const v8bf*)(Wt + (size_t)(n0 + r) * K + kn + sc);
    }
  }

  // ---- compute from buffer CUR while the loads above are in flight ----
  // A fragment: lanes 0-15 K {0..7,16..23}; lanes 16-31 K {8..15,24..31}
  v16bf afr[2], bfr[4];
  #pragma unroll
  for (int tm = 0; tm < 2; ++tm) {
    int row = waveM * 32 + tm * 16 + laneLo;
    v8bf lo = *(const v8bf*)&As[CUR][row][laneHi * 8];
    v8bf hi = *(const v8bf*)&As[CUR][row][16 + laneHi * 8];
    afr[tm] = __builtin_shufflevector(lo, hi, 0,1,2,3,4,5,6,7,8,9,10,11,12,13,14,15);
  }
  // B fragment: lanes 0-15 K {0..7,16..23}; lanes 16-31 K {8..15,24..31}; N = laneLo
  #pragma unroll
  for (int tn = 0; tn < 4; ++tn) {
    int col = waveN * 64 + tn * 16 + laneLo;
    v8bf lo = *(const v8bf*)&Bs[CUR][col][laneHi * 8];
    v8bf hi = *(const v8bf*)&Bs[CUR][col][16 + laneHi * 8];
    bfr[tn] = __builtin_shufflevector(lo, hi, 0,1,2,3,4,5,6,7,8,9,10,11,12,13,14,15);
  }
  #pragma unroll
  for (int tm = 0; tm < 2; ++tm)
    #pragma unroll
    for (int tn = 0; tn < 4; ++tn)
      acc[tm][tn] = __builtin_amdgcn_wmma_f32_16x16x32_bf16(
          false, afr[tm], false, bfr[tn], (short)0, acc[tm][tn], false, false);

  // ---- commit staged tile into the other buffer ----
  if (have) {
    #pragma unroll
    for (int j = 0; j < 2; ++j) {
      int r = j * 64 + sr;
      unsigned p0, p1, p2, p3;
      if (okA[j]) {
        float4 f0 = fa[j][0], f1 = fa[j][1];
        if (GATED) {
          float4 g0 = fg[j][0], g1 = fg[j][1];
          f0.x *= g0.x; f0.y *= g0.y; f0.z *= g0.z; f0.w *= g0.w;
          f1.x *= g1.x; f1.y *= g1.y; f1.z *= g1.z; f1.w *= g1.w;
        }
        p0 = pack2bf(f0.x, f0.y); p1 = pack2bf(f0.z, f0.w);
        p2 = pack2bf(f1.x, f1.y); p3 = pack2bf(f1.z, f1.w);
      } else {
        p0 = p1 = p2 = p3 = 0u;
      }
      *(uint4*)&As[CUR ^ 1][r][sc] = make_uint4(p0, p1, p2, p3);
      *(v8bf*)&Bs[CUR ^ 1][r][sc] = fb[j];
    }
  }
}

// ---------------------- Generic WMMA GEMM (bf16 MAC, f32 acc) --------------
// Y[m, ycoloff+n] = A[m,:K] @ W[:K,n] + bias[n], W given as bf16 Wt[N][K].
// Block tile 128x128, 8 waves, each wave 32(M) x 64(N) = 2x4 WMMA tiles.
template <bool GATED, bool BOUNDS>
__global__ __launch_bounds__(256)
void gemm_bias_wmma(const float* __restrict__ A, int lda,
                    const __bf16* __restrict__ Wt,
                    const float* __restrict__ bias,
                    float* __restrict__ Y, int ldy, int ycoloff,
                    const float* __restrict__ gate, int rows_per_batch,
                    int M, int K, int N) {
  __shared__ alignas(16) __bf16 As[2][BM][LDP];   // row-major [m][k]
  __shared__ alignas(16) __bf16 Bs[2][BN][LDP];   // transposed [n][k]

  const int tid    = threadIdx.x;
  const int wave   = tid >> 5;
  const int lane   = tid & 31;
  const int waveM  = wave >> 1;        // 0..3  (32 rows each)
  const int waveN  = wave & 1;         // 0..1  (64 cols each)
  const int laneLo = lane & 15;
  const int laneHi = lane >> 4;
  const int sr     = tid >> 2;         // staging row within half-tile (0..63)
  const int sc     = (tid & 3) * 8;    // staging k offset (0/8/16/24)

  const int n0 = blockIdx.x * BN;
  const int m0 = blockIdx.y * BM;

  const int nk = K / BK;
  __builtin_assume(nk >= 1);

  v8f zero = {};
  v8f acc[2][4];
  #pragma unroll
  for (int i = 0; i < 2; ++i)
    #pragma unroll
    for (int j = 0; j < 4; ++j) acc[i][j] = zero;

  // ---- prologue: stage k-tile 0 into buffer 0 ----
  #pragma unroll
  for (int j = 0; j < 2; ++j) {
    int r = j * 64 + sr;
    int gm = m0 + r;
    unsigned p0, p1, p2, p3;
    if (!BOUNDS || gm < M) {
      const float* ap = A + (size_t)gm * lda + sc;
      float4 f0 = *(const float4*)ap;
      float4 f1 = *(const float4*)(ap + 4);
      if (GATED) {
        const float* gp = gate + (size_t)(gm / rows_per_batch) * K + sc;
        float4 g0 = *(const float4*)gp;
        float4 g1 = *(const float4*)(gp + 4);
        f0.x *= g0.x; f0.y *= g0.y; f0.z *= g0.z; f0.w *= g0.w;
        f1.x *= g1.x; f1.y *= g1.y; f1.z *= g1.z; f1.w *= g1.w;
      }
      p0 = pack2bf(f0.x, f0.y); p1 = pack2bf(f0.z, f0.w);
      p2 = pack2bf(f1.x, f1.y); p3 = pack2bf(f1.z, f1.w);
    } else {
      p0 = p1 = p2 = p3 = 0u;
    }
    *(uint4*)&As[0][r][sc] = make_uint4(p0, p1, p2, p3);
    *(v8bf*)&Bs[0][r][sc] = *(const v8bf*)(Wt + (size_t)(n0 + r) * K + sc);
  }

  // ---- main pipeline: unrolled x2 so the LDS buffer index is constant ----
  for (int it = 0; it < nk; it += 2) {
    gemm_step<GATED, BOUNDS, 0>(A, lda, Wt, gate, rows_per_batch, M, K, it, nk,
                                m0, n0, sr, sc, waveM, waveN, laneLo, laneHi,
                                As, Bs, acc);
    if (it + 1 < nk)
      gemm_step<GATED, BOUNDS, 1>(A, lda, Wt, gate, rows_per_batch, M, K, it + 1, nk,
                                  m0, n0, sr, sc, waveM, waveN, laneLo, laneHi,
                                  As, Bs, acc);
  }

  // C/D layout: vgpr r -> M = r + 8*laneHi, N = laneLo
  #pragma unroll
  for (int tm = 0; tm < 2; ++tm) {
    #pragma unroll
    for (int tn = 0; tn < 4; ++tn) {
      int gn = n0 + waveN * 64 + tn * 16 + laneLo;
      float bv = bias[gn];
      #pragma unroll
      for (int r = 0; r < 8; ++r) {
        int gm = m0 + waveM * 32 + tm * 16 + laneHi * 8 + r;
        if (!BOUNDS || gm < M)
          Y[(size_t)gm * ldy + ycoloff + gn] = acc[tm][tn][r] + bv;
      }
    }
  }
}

// ------------------- Depthwise 7x7 conv (SAME), 4 ch/thread ----------------
__global__ void dwconv7_kernel(const float* __restrict__ in, int ldin,
                               const float* __restrict__ w, const float* __restrict__ bias,
                               float* __restrict__ out, int ldout, int outoff,
                               int B, int H, int Wd, int C) {
  int C4 = C >> 2;
  size_t idx = (size_t)blockIdx.x * blockDim.x + threadIdx.x;
  size_t total = (size_t)B * H * Wd * C4;
  if (idx >= total) return;
  int c = (int)(idx % C4) * 4;
  size_t t = idx / C4;
  int xx = (int)(t % Wd);
  int yy = (int)((t / Wd) % H);
  int b  = (int)(t / ((size_t)Wd * H));
  float4 acc = *(const float4*)&bias[c];
  for (int ky = 0; ky < 7; ++ky) {
    int iy = yy + ky - 3;
    if (iy < 0 || iy >= H) continue;
    #pragma unroll
    for (int kx = 0; kx < 7; ++kx) {
      int ix = xx + kx - 3;
      if (ix < 0 || ix >= Wd) continue;
      size_t tt = ((size_t)b * H + iy) * Wd + ix;
      float4 vi = *(const float4*)&in[tt * ldin + c];
      float4 wv = *(const float4*)&w[(ky * 7 + kx) * C + c];
      acc.x += vi.x * wv.x; acc.y += vi.y * wv.y;
      acc.z += vi.z * wv.z; acc.w += vi.w * wv.w;
    }
  }
  *(float4*)&out[t * ldout + outoff + c] = acc;
}

// ----------------- local_rr gate: CAT[:,0:256] = T1 * T2 -------------------
__global__ void rrgate_kernel(const float* __restrict__ a, const float* __restrict__ b,
                              float* __restrict__ cat, size_t Mtok) {
  size_t idx = (size_t)blockIdx.x * 256 + threadIdx.x;   // float4 units
  if (idx >= Mtok * 64) return;
  size_t t = idx >> 6; int c = (int)(idx & 63) * 4;
  float4 av = *(const float4*)&a[t * 256 + c];
  float4 bv = *(const float4*)&b[t * 256 + c];
  float4 o; o.x = av.x * bv.x; o.y = av.y * bv.y; o.z = av.z * bv.z; o.w = av.w * bv.w;
  *(float4*)&cat[t * 512 + c] = o;
}

// --------------- rgb*t and |rgb - t| (inputs to lx convs) ------------------
__global__ void proddiff_kernel(const float* __restrict__ t2, float* __restrict__ t1,
                                size_t Mtok) {
  size_t idx = (size_t)blockIdx.x * 256 + threadIdx.x;   // float4 units
  if (idx >= Mtok * 32) return;
  size_t e = idx * 4;
  float4 r = *(const float4*)&t2[e];
  float4 te = *(const float4*)&t2[Mtok * 128 + e];
  float4 p, d;
  p.x = r.x * te.x; p.y = r.y * te.y; p.z = r.z * te.z; p.w = r.w * te.w;
  d.x = fabsf(r.x - te.x); d.y = fabsf(r.y - te.y);
  d.z = fabsf(r.z - te.z); d.w = fabsf(r.w - te.w);
  *(float4*)&t1[e] = p;
  *(float4*)&t1[Mtok * 128 + e] = d;
}

__global__ void zero_kernel(float* __restrict__ p, int n) {
  int i = blockIdx.x * 256 + threadIdx.x;
  if (i < n) p[i] = 0.f;
}

// ----------- channel-gate statistics: amap / dot / n1^2 / n2^2 -------------
__global__ __launch_bounds__(256)
void gate_stats_kernel(const float* __restrict__ codi, float* __restrict__ n1sq,
                       float* __restrict__ dot, float* __restrict__ n2sq,
                       int hw, int tpc) {
  __shared__ float sm[256];
  int b = blockIdx.x, chunk = blockIdx.y, c = threadIdx.x;
  int t0 = chunk * tpc;
  float dAcc = 0.f, n2Acc = 0.f, aAcc = 0.f;
  for (int i = 0; i < tpc; ++i) {
    size_t t = (size_t)b * hw + t0 + i;
    float v = codi[t * 256 + c];
    sm[c] = v; __syncthreads();
    for (int s = 128; s > 0; s >>= 1) { if (c < s) sm[c] += sm[c + s]; __syncthreads(); }
    float amap = sm[0] * (1.0f / 256.0f);
    __syncthreads();
    dAcc += amap * v;
    n2Acc += v * v;
    if (c == 0) aAcc += amap * amap;
  }
  atomicAdd(&dot[b * 256 + c], dAcc);
  atomicAdd(&n2sq[b * 256 + c], n2Acc);
  if (c == 0) atomicAdd(&n1sq[b], aAcc);
}

// ------------- cos_sim -> fc1 -> gelu -> fc2 -> sigmoid gate ---------------
__global__ void gate_fc_kernel(const float* __restrict__ n1sq, const float* __restrict__ dot,
                               const float* __restrict__ n2sq, const float* __restrict__ fc1,
                               const float* __restrict__ fc2, float* __restrict__ attnc) {
  __shared__ float cs[256];
  __shared__ float h[32];
  int b = blockIdx.x, c = threadIdx.x;
  float n1 = sqrtf(n1sq[b]);
  cs[c] = dot[b * 256 + c] / (n1 * sqrtf(n2sq[b * 256 + c]) + 1e-6f);
  __syncthreads();
  if (c < 32) {
    float s = 0.f;
    for (int k = 0; k < 256; ++k) s += cs[k] * fc1[k * 32 + c];
    h[c] = 0.5f * s * (1.0f + erff(s * 0.70710678118654752f));  // exact GELU
  }
  __syncthreads();
  float s2 = 0.f;
  #pragma unroll
  for (int j = 0; j < 32; ++j) s2 += h[j] * fc2[j * 256 + c];
  attnc[b * 256 + c] = 1.0f / (1.0f + __expf(-s2));
}

// ----------------- 8x8 average pool of concat(x, x_e) ----------------------
__global__ void rxpool_kernel(const float* __restrict__ xln, const float* __restrict__ xeln,
                              float* __restrict__ pool) {
  int bp = blockIdx.x;               // b*49 + p
  int b = bp / 49, p = bp % 49;
  int py = p / 7, px = p % 7;
  for (int c = threadIdx.x; c < 384; c += blockDim.x) {
    float s = 0.f;
    for (int iy = 0; iy < 8; ++iy)
      for (int ix = 0; ix < 8; ++ix) {
        size_t t = ((size_t)b * 56 + py * 8 + iy) * 56 + px * 8 + ix;
        s += (c < 256) ? xln[t * 256 + c] : xeln[t * 128 + (c - 256)];
      }
    pool[(size_t)bp * 384 + c] = s * (1.0f / 64.0f);
  }
}

// --------------- pooled-q attention: one block per (b, head, q) ------------
__global__ __launch_bounds__(256)
void attn_kernel(const float* __restrict__ Q, const float* __restrict__ KV,
                 float* __restrict__ gsmall) {
  __shared__ float sc[3136];
  __shared__ float red[256];
  __shared__ float4 qv[4];
  __shared__ float accD[16];
  int qi = blockIdx.x, n = blockIdx.y, b = blockIdx.z;
  int tid = threadIdx.x;
  if (tid < 4) {
    float4 q4 = *(const float4*)&Q[((size_t)(b * 49 + qi)) * 128 + n * 16 + tid * 4];
    q4.x *= 0.25f; q4.y *= 0.25f; q4.z *= 0.25f; q4.w *= 0.25f;   // hd^-0.5
    qv[tid] = q4;
  }
  if (tid < 16) accD[tid] = 0.f;
  __syncthreads();
  float4 q0 = qv[0], q1 = qv[1], q2 = qv[2], q3 = qv[3];
  float lmax = -1e30f;
  for (int t = tid; t < 3136; t += 256) {
    const float4* kp = (const float4*)&KV[((size_t)(b * 3136 + t)) * 256 + n * 16];
    float4 k0 = kp[0], k1 = kp[1], k2 = kp[2], k3 = kp[3];
    float s = q0.x * k0.x + q0.y * k0.y + q0.z * k0.z + q0.w * k0.w
            + q1.x * k1.x + q1.y * k1.y + q1.z * k1.z + q1.w * k1.w
            + q2.x * k2.x + q2.y * k2.y + q2.z * k2.z + q2.w * k2.w
            + q3.x * k3.x + q3.y * k3.y + q3.z * k3.z + q3.w * k3.w;
    sc[t] = s;
    lmax = fmaxf(lmax, s);
  }
  red[tid] = lmax; __syncthreads();
  for (int s = 128; s > 0; s >>= 1) { if (tid < s) red[tid] = fmaxf(red[tid], red[tid + s]); __syncthreads(); }
  float m = red[0];
  __syncthreads();
  float lsum = 0.f;
  float4 a0 = {}, a1 = {}, a2 = {}, a3 = {};
  for (int t = tid; t < 3136; t += 256) {
    float p = __expf(sc[t] - m);
    lsum += p;
    const float4* vp = (const float4*)&KV[((size_t)(b * 3136 + t)) * 256 + 128 + n * 16];
    float4 v0 = vp[0], v1 = vp[1], v2 = vp[2], v3 = vp[3];
    a0.x += p * v0.x; a0.y += p * v0.y; a0.z += p * v0.z; a0.w += p * v0.w;
    a1.x += p * v1.x; a1.y += p * v1.y; a1.z += p * v1.z; a1.w += p * v1.w;
    a2.x += p * v2.x; a2.y += p * v2.y; a2.z += p * v2.z; a2.w += p * v2.w;
    a3.x += p * v3.x; a3.y += p * v3.y; a3.z += p * v3.z; a3.w += p * v3.w;
  }
  red[tid] = lsum; __syncthreads();
  for (int s = 128; s > 0; s >>= 1) { if (tid < s) red[tid] += red[tid + s]; __syncthreads(); }
  float sum = red[0];
  float la[16] = { a0.x, a0.y, a0.z, a0.w, a1.x, a1.y, a1.z, a1.w,
                   a2.x, a2.y, a2.z, a2.w, a3.x, a3.y, a3.z, a3.w };
  #pragma unroll
  for (int d = 0; d < 16; ++d) atomicAdd(&accD[d], la[d]);
  __syncthreads();
  if (tid < 16)
    gsmall[((size_t)(b * 128) + n * 16 + tid) * 49 + qi] = accD[tid] / sum;
}

// ------------- bilinear 7x7 -> 56x56 (half-pixel centers) ------------------
__global__ void upsample_kernel(const float* __restrict__ gsmall, float* __restrict__ cat) {
  int t = blockIdx.x;                 // token over B*56*56
  int c = threadIdx.x;                // 128 channels
  int xx = t % 56, yy = (t / 56) % 56, b = t / 3136;
  float sy = (yy + 0.5f) * 0.125f - 0.5f;
  float sx = (xx + 0.5f) * 0.125f - 0.5f;
  int y0 = (int)floorf(sy), x0 = (int)floorf(sx);
  float wy = sy - (float)y0, wx = sx - (float)x0;
  int y0c = min(max(y0, 0), 6), y1c = min(max(y0 + 1, 0), 6);
  int x0c = min(max(x0, 0), 6), x1c = min(max(x0 + 1, 0), 6);
  const float* gp = &gsmall[((size_t)(b * 128) + c) * 49];
  float v = (1.f - wy) * ((1.f - wx) * gp[y0c * 7 + x0c] + wx * gp[y0c * 7 + x1c])
          +        wy  * ((1.f - wx) * gp[y1c * 7 + x0c] + wx * gp[y1c * 7 + x1c]);
  cat[(size_t)t * 512 + 384 + c] = v;
}

// ---------------------------------------------------------------------------
extern "C" void kernel_launch(void* const* d_in, const int* in_sizes, int n_in,
                              void* d_out, int out_size, void* d_ws, size_t ws_size,
                              hipStream_t stream) {
  (void)in_sizes; (void)n_in; (void)out_size; (void)ws_size;
  const int B = 16, H = 56, Wd = 56;
  const size_t M = (size_t)B * H * Wd;          // 50176 tokens

  const float* x       = (const float*)d_in[0];
  const float* x_e     = (const float*)d_in[1];
  const float* norm_w  = (const float*)d_in[2];
  const float* norm_b  = (const float*)d_in[3];
  const float* norme_w = (const float*)d_in[4];
  const float* norme_b = (const float*)d_in[5];
  const float* rr1_w   = (const float*)d_in[6];
  const float* rr1_b   = (const float*)d_in[7];
  const float* rr2_w   = (const float*)d_in[8];
  const float* rr2_b   = (const float*)d_in[9];
  const float* rr3_w   = (const float*)d_in[10];
  const float* rr3_b   = (const float*)d_in[11];
  const float* conv_w  = (const float*)d_in[12];
  const float* conv_b  = (const float*)d_in[13];
  const float* l1_w    = (const float*)d_in[14];
  const float* l1_b    = (const float*)d_in[15];
  const float* l2_w    = (const float*)d_in[16];
  const float* l2_b    = (const float*)d_in[17];
  const float* c1_w    = (const float*)d_in[18];
  const float* c1_b    = (const float*)d_in[19];
  const float* c2_w    = (const float*)d_in[20];
  const float* c2_b    = (const float*)d_in[21];
  const float* fc1_w   = (const float*)d_in[22];
  const float* fc2_w   = (const float*)d_in[23];
  const float* l3_w    = (const float*)d_in[24];
  const float* l3_b    = (const float*)d_in[25];
  const float* kv_w    = (const float*)d_in[26];
  const float* kv_b    = (const float*)d_in[27];
  const float* q_w     = (const float*)d_in[28];
  const float* q_b     = (const float*)d_in[29];
  const float* proj_w  = (const float*)d_in[30];
  const float* proj_b  = (const float*)d_in[31];
  const float* proje_w = (const float*)d_in[32];
  const float* proje_b = (const float*)d_in[33];
  float* out = (float*)d_out;

  // ----- workspace layout (floats) -----
  float* ws     = (float*)d_ws;
  float* X_LN   = ws;                  // M*256
  float* XE_LN  = X_LN  + M * 256;     // M*128
  float* T1     = XE_LN + M * 128;     // M*256 scratch
  float* T2     = T1    + M * 256;     // M*256 scratch
  float* CAT    = T2    + M * 256;     // M*512 concat buffer
  float* N1SQ   = CAT   + M * 512;     // 16
  float* DOT    = N1SQ  + 16;          // 16*256
  float* N2SQ   = DOT   + 16 * 256;    // 16*256
  float* ATTNC  = N2SQ  + 16 * 256;    // 16*256
  float* RXPOOL = ATTNC + 16 * 256;    // 784*384
  float* QBUF   = RXPOOL + 784 * 384;  // 784*128
  float* GSMALL = QBUF  + 784 * 128;   // 16*128*49

  // bf16 transposed weight cache ([N][K] halves)
  __bf16* WTB    = (__bf16*)(GSMALL + 16 * 128 * 49);
  __bf16* rr2t   = WTB;                  // 256*256
  __bf16* rr3t   = rr2t  + 65536;        // 256*256
  __bf16* rr1t   = rr3t  + 65536;        // 256*256
  __bf16* l1t    = rr1t  + 65536;        // 128*256
  __bf16* l2t    = l1t   + 32768;        // 128*128
  __bf16* l3t    = l2t   + 16384;        // 128*256
  __bf16* kvt    = l3t   + 32768;        // 256*256
  __bf16* qt     = kvt   + 65536;        // 128*384
  __bf16* projt  = qt    + 49152;        // 256*512
  __bf16* projet = projt + 131072;       // 128*512

  dim3 blk(256);

  // ----- one-time (per launch) weight convert+transpose into bf16 cache ----
  wconv_kernel<<<128, blk, 0, stream>>>(rr2_w,  (unsigned*)rr2t,  256, 256);
  wconv_kernel<<<128, blk, 0, stream>>>(rr3_w,  (unsigned*)rr3t,  256, 256);
  wconv_kernel<<<128, blk, 0, stream>>>(rr1_w,  (unsigned*)rr1t,  256, 256);
  wconv_kernel<<< 64, blk, 0, stream>>>(l1_w,   (unsigned*)l1t,   256, 128);
  wconv_kernel<<< 32, blk, 0, stream>>>(l2_w,   (unsigned*)l2t,   128, 128);
  wconv_kernel<<< 64, blk, 0, stream>>>(l3_w,   (unsigned*)l3t,   256, 128);
  wconv_kernel<<<128, blk, 0, stream>>>(kv_w,   (unsigned*)kvt,   256, 256);
  wconv_kernel<<< 96, blk, 0, stream>>>(q_w,    (unsigned*)qt,    384, 128);
  wconv_kernel<<<256, blk, 0, stream>>>(proj_w, (unsigned*)projt, 512, 256);
  wconv_kernel<<<128, blk, 0, stream>>>(proje_w,(unsigned*)projet,512, 128);

  const int redN = 16 + 3 * 16 * 256;
  zero_kernel<<<(redN + 255) / 256, 256, 0, stream>>>(N1SQ, redN);

  // ----- LayerNorms -----
  ln_kernel<<<(int)M, 256, 0, stream>>>(x, norm_w, norm_b, X_LN, 256);
  ln_kernel<<<(int)M, 128, 0, stream>>>(x_e, norme_w, norme_b, XE_LN, 128);

  dim3 gN256(2, (unsigned)(M / BM));   // N=256 output
  dim3 gN128(1, (unsigned)(M / BM));   // N=128 output

  // ----- local_rr: rr1 * (dwconv(rr2) @ rr3) -----
  gemm_bias_wmma<false, false><<<gN256, blk, 0, stream>>>(
      X_LN, 256, rr2t, rr2_b, T1, 256, 0, nullptr, 1, (int)M, 256, 256);
  dwconv7_kernel<<<(int)((M * 64 + 255) / 256), blk, 0, stream>>>(
      T1, 256, conv_w, conv_b, T2, 256, 0, B, H, Wd, 256);
  gemm_bias_wmma<false, false><<<gN256, blk, 0, stream>>>(
      T2, 256, rr3t, rr3_b, T1, 256, 0, nullptr, 1, (int)M, 256, 256);
  gemm_bias_wmma<false, false><<<gN256, blk, 0, stream>>>(
      X_LN, 256, rr1t, rr1_b, T2, 256, 0, nullptr, 1, (int)M, 256, 256);
  rrgate_kernel<<<(int)((M * 64 + 255) / 256), blk, 0, stream>>>(T1, T2, CAT, M);

  // ----- LocalAttentionRGBT -----
  gemm_bias_wmma<false, false><<<gN128, blk, 0, stream>>>(
      X_LN, 256, l1t, l1_b, T2, 128, 0, nullptr, 1, (int)M, 256, 128);            // rgb
  gemm_bias_wmma<false, false><<<gN128, blk, 0, stream>>>(
      XE_LN, 128, l2t, l2_b, T2 + M * 128, 128, 0, nullptr, 1, (int)M, 128, 128); // t
  proddiff_kernel<<<(int)((M * 32 + 255) / 256), blk, 0, stream>>>(T2, T1, M);
  dwconv7_kernel<<<(int)((M * 32 + 255) / 256), blk, 0, stream>>>(
      T1, 128, c1_w, c1_b, T2, 256, 0, B, H, Wd, 128);                            // co
  dwconv7_kernel<<<(int)((M * 32 + 255) / 256), blk, 0, stream>>>(
      T1 + M * 128, 128, c2_w, c2_b, T2, 256, 128, B, H, Wd, 128);                // di
  gate_stats_kernel<<<dim3(16, 28), blk, 0, stream>>>(T2, N1SQ, DOT, N2SQ, 3136, 112);
  gate_fc_kernel<<<16, blk, 0, stream>>>(N1SQ, DOT, N2SQ, fc1_w, fc2_w, ATTNC);
  gemm_bias_wmma<true, false><<<gN128, blk, 0, stream>>>(
      T2, 256, l3t, l3_b, CAT, 512, 256, ATTNC, 3136, (int)M, 256, 128);          // local_rx

  // ----- global pooled-q attention -----
  gemm_bias_wmma<false, false><<<gN256, blk, 0, stream>>>(
      X_LN, 256, kvt, kv_b, T1, 256, 0, nullptr, 1, (int)M, 256, 256);            // kv
  rxpool_kernel<<<16 * 49, blk, 0, stream>>>(X_LN, XE_LN, RXPOOL);
  gemm_bias_wmma<false, true><<<dim3(1, 7), blk, 0, stream>>>(
      RXPOOL, 384, qt, q_b, QBUF, 128, 0, nullptr, 1, 784, 384, 128);             // q
  attn_kernel<<<dim3(49, 8, 16), blk, 0, stream>>>(QBUF, T1, GSMALL);
  upsample_kernel<<<(int)M, 128, 0, stream>>>(GSMALL, CAT);

  // ----- output projections from concat [local_rr | local_rx | g] -----
  gemm_bias_wmma<false, false><<<gN256, blk, 0, stream>>>(
      CAT, 512, projt, proj_b, out, 256, 0, nullptr, 1, (int)M, 512, 256);
  gemm_bias_wmma<false, false><<<gN128, blk, 0, stream>>>(
      CAT, 512, projet, proje_b, out + M * 256, 128, 0, nullptr, 1, (int)M, 512, 128);
}
